// DynamicMultiHeadConv_89653147337265
// MI455X (gfx1250) — matrix operand order repaired and checked by reading, not compiled, not run on hardware
//
#include <hip/hip_runtime.h>
#include <stdint.h>

// Problem constants (from reference)
#define B_     4
#define C_     96
#define S_     24
#define H_     32
#define W_     32
#define HEADS_ 4
#define COUT_  192
#define MH_    48              // COUT / HEADS
#define HID_   6
#define K3_    27              // 3*3*3
#define KDIM_  (C_ * K3_)      // 2592
#define NTOT_  (S_ * H_ * W_)  // 24576
#define HW_    (H_ * W_)
#define BN_EPS_ 1e-5f

// Conv tiling
#define NTILE_ 64              // N (spatial) per block
#define KCH_   32              // K per WMMA step (f16 16x16x32)
#define BSTR_  40              // LDS row stride in halves (32 + 8 pad)
#define KTAB_  2688            // padded compacted-K capacity (84 chunks * 32)

typedef _Float16 half_t;
typedef half_t v16h __attribute__((ext_vector_type(16)));
typedef float  v8f  __attribute__((ext_vector_type(8)));

union Frag {
    v16h v;
    uint4 q[2];
};

// ---------------------------------------------------------------------------
// Kernel 1: BatchNorm(eval) + ReLU -> f16 xr, plus global average pool -> xa
// One block per (b, c).
// ---------------------------------------------------------------------------
__global__ __launch_bounds__(256) void k_bn_relu_pool(
    const float* __restrict__ x,
    const float* __restrict__ gamma, const float* __restrict__ beta,
    const float* __restrict__ mean,  const float* __restrict__ var,
    half_t* __restrict__ xr, float* __restrict__ xa)
{
    int bc = blockIdx.x;            // b*C_ + c
    int c  = bc % C_;
    float sc = gamma[c] * rsqrtf(var[c] + BN_EPS_);
    float sh = beta[c] - mean[c] * sc;

    const float* xin = x + (size_t)bc * NTOT_;
    half_t* xo = xr + (size_t)bc * NTOT_;

    float sum = 0.f;
    for (int i = threadIdx.x; i < NTOT_; i += 256) {
        float v = fmaxf(xin[i] * sc + sh, 0.f);
        sum += v;
        xo[i] = (half_t)v;
    }
    __shared__ float red[256];
    red[threadIdx.x] = sum;
    __syncthreads();
    for (int off = 128; off > 0; off >>= 1) {
        if (threadIdx.x < off) red[threadIdx.x] += red[threadIdx.x + off];
        __syncthreads();
    }
    if (threadIdx.x == 0) xa[bc] = red[0] * (1.0f / NTOT_);
}

// ---------------------------------------------------------------------------
// Kernel 2: SE gate MLP + top-k (smallest) pruning + deterministic
// compaction of active channels. One block per (b, head).
// ---------------------------------------------------------------------------
__global__ __launch_bounds__(128) void k_gate(
    const float* __restrict__ xa,
    const float* __restrict__ fc1,   // [HEADS][HID][C]
    const float* __restrict__ fc2,   // [HEADS][C][HID]
    const float* __restrict__ fc2b,  // [HEADS][C]
    const int*   __restrict__ inact,
    float* __restrict__ mask,        // [B][HEADS][C]
    int*   __restrict__ actList,     // [B][HEADS][C]
    int*   __restrict__ actCnt)      // [B][HEADS]
{
    int bh = blockIdx.x;             // b*HEADS + h
    int b  = bh / HEADS_;
    int h  = bh % HEADS_;
    int t  = threadIdx.x;

    __shared__ float ylds[HID_];
    __shared__ float md[C_];
    __shared__ float thr;

    if (t < HID_) {
        const float* w1 = fc1 + ((size_t)h * HID_ + t) * C_;
        const float* xv = xa + b * C_;
        float s = 0.f;
        for (int c = 0; c < C_; ++c) s += xv[c] * w1[c];
        ylds[t] = fmaxf(s, 0.f);
    }
    __syncthreads();
    if (t < C_) {
        const float* w2 = fc2 + ((size_t)h * C_ + t) * HID_;
        float s = fc2b[h * C_ + t];
        for (int r = 0; r < HID_; ++r) s += ylds[r] * w2[r];
        md[t] = fmaxf(s, 0.f);
    }
    __syncthreads();

    int ic = inact[0];
    if (ic > C_) ic = C_;
    if (ic > 0) {
        // k-th smallest (sorted index ic-1) by rank counting; ties share value
        if (t < C_) {
            float v = md[t];
            int lt = 0, le = 0;
            for (int j = 0; j < C_; ++j) {
                lt += (md[j] <  v) ? 1 : 0;
                le += (md[j] <= v) ? 1 : 0;
            }
            if (lt <= (ic - 1) && (ic - 1) < le) thr = v;  // all writers agree
        }
        __syncthreads();
        if (t < C_) md[t] = (md[t] <= thr) ? 0.f : md[t];
        __syncthreads();
    }

    if (t < C_) {
        float mv = md[t];
        mask[(size_t)bh * C_ + t] = mv;
        // deterministic compaction: pos = #active with smaller index
        int pos = 0, tot = 0;
        for (int j = 0; j < C_; ++j) {
            int a = (md[j] > 0.f) ? 1 : 0;
            tot += a;
            if (j < t) pos += a;
        }
        if (mv > 0.f) actList[(size_t)bh * C_ + pos] = t;
        if (t == 0)   actCnt[bh] = tot;
    }
}

// ---------------------------------------------------------------------------
// Kernel 3: build compacted mask-scaled f16 A matrix Apack[bh][48][KTAB]
// (zero-padded beyond Kact). One-time gather; conv then streams it.
// ---------------------------------------------------------------------------
__global__ __launch_bounds__(256) void k_prepA(
    const float* __restrict__ convw,  // [HEADS][MH][C][27]
    const float* __restrict__ mask,   // [B][HEADS][C]
    const int*   __restrict__ actList,
    const int*   __restrict__ actCnt,
    half_t* __restrict__ Apack)
{
    int i = blockIdx.x * blockDim.x + threadIdx.x;
    if (i >= B_ * HEADS_ * MH_ * KTAB_) return;
    int bh  = i / (MH_ * KTAB_);
    int rem = i - bh * (MH_ * KTAB_);
    int m   = rem / KTAB_;
    int k   = rem - m * KTAB_;
    int head = bh & (HEADS_ - 1);

    int cnt  = actCnt[bh];
    int Kact = cnt * K3_;
    half_t v = (half_t)0.f;
    if (k < Kact) {
        int ci = k / K3_;
        int ko = k - ci * K3_;
        int c  = actList[(size_t)bh * C_ + ci];
        float wv = convw[(((size_t)(head * MH_ + m)) * C_ + c) * K3_ + ko];
        v = (half_t)(wv * mask[(size_t)bh * C_ + c]);
    }
    Apack[i] = v;
}

// ---------------------------------------------------------------------------
// Kernel 4: build per-(bh, k) im2col decode table:
//   .x = xr offset delta  (c*S*HW + (kd-1)*HW + (kh-1)*W + (kw-1))
//   .y = packed kd | kh<<8 | kw<<16   (for branchless bounds check)
// Padding entries -> channel 0, kernel center (always in-bounds; A is 0).
// ---------------------------------------------------------------------------
__global__ __launch_bounds__(256) void k_prepT(
    const int* __restrict__ actList,
    const int* __restrict__ actCnt,
    int2* __restrict__ tabG)
{
    int i = blockIdx.x * blockDim.x + threadIdx.x;
    if (i >= B_ * HEADS_ * KTAB_) return;
    int bh = i / KTAB_;
    int k  = i - bh * KTAB_;

    int cnt  = actCnt[bh];
    int Kact = cnt * K3_;
    int c = 0, kd = 1, kh = 1, kw = 1;
    if (k < Kact) {
        int ci = k / K3_;
        int ko = k - ci * K3_;
        c  = actList[(size_t)bh * C_ + ci];
        kd = ko / 9;
        int kr = ko - kd * 9;
        kh = kr / 3;
        kw = kr - kh * 3;
    }
    int2 e;
    e.x = c * (S_ * HW_) + (kd - 1) * HW_ + (kh - 1) * W_ + (kw - 1);
    e.y = kd | (kh << 8) | (kw << 16);
    tabG[i] = e;
}

// ---------------------------------------------------------------------------
// Kernel 5: implicit-GEMM grouped conv3d with WMMA f16 over COMPACTED K.
// Grid (NTOT/64, HEADS, B), 96 threads = 3 waves; wave w owns M-rows
// [16w,16w+16). A fragments stream from L2-resident Apack (2 global b128s);
// B tile is a branchless register-table im2col gather into LDS.
// Channel shuffle folded into the output index.
// ---------------------------------------------------------------------------
__global__ __launch_bounds__(96) void k_conv(
    const half_t* __restrict__ xr,     // [B][C][S][H][W] f16 (post BN+ReLU)
    const half_t* __restrict__ Apack,  // [BH][MH][KTAB] f16
    const int2*   __restrict__ tabG,   // [BH][KTAB]
    const int*    __restrict__ actCnt,
    float* __restrict__ out)           // [B][COUT][S][H][W]
{
    int nblock = blockIdx.x;
    int head   = blockIdx.y;
    int b      = blockIdx.z;
    int bh     = b * HEADS_ + head;
    int tid    = threadIdx.x;
    int lane   = tid & 31;
    int wave   = tid >> 5;        // 0..2 : M-tile
    int hi     = lane >> 4;       // lane group (0: lanes 0-15, 1: lanes 16-31)
    int lan    = lane & 15;
    int kk     = tid & 31;        // fixed B-column per thread (96 % 32 == 0)

    __shared__ __align__(16) half_t Bs[NTILE_ * BSTR_];

    int cnt    = actCnt[bh];
    int chunks = (cnt * K3_ + KCH_ - 1) >> 5;

    const half_t* xb   = xr + (size_t)b * C_ * NTOT_;
    const half_t* arow = Apack + ((size_t)bh * MH_ + wave * 16 + lan) * KTAB_;
    const int2*   tab  = tabG + (size_t)bh * KTAB_ + kk;
    int n0 = nblock * NTILE_;

    v8f acc0 = {}, acc1 = {}, acc2 = {}, acc3 = {};

    for (int kc = 0; kc < chunks; ++kc) {
        __syncthreads();   // protect previous iteration's fragment reads

        // --- B tile: im2col gather, decode entry held in registers ---
        int2 ent = tab[kc * KCH_];
        int kd = ent.y & 0xff;
        int kh = (ent.y >> 8) & 0xff;
        int kw = (ent.y >> 16) & 0xff;
        for (int i = tid; i < NTILE_ * KCH_; i += 96) {
            int nl = i >> 5;
            int n  = n0 + nl;
            int s  = n >> 10;
            int hh = (n >> 5) & 31;
            int ww = n & 31;
            bool ok = ((unsigned)(s + kd - 1) < (unsigned)S_) &&
                      ((unsigned)(hh + kh - 1) < (unsigned)H_) &&
                      ((unsigned)(ww + kw - 1) < (unsigned)W_);
            int off = ent.x + n;
            half_t v = xb[ok ? off : 0];          // cndmask'd safe address
            Bs[nl * BSTR_ + kk] = ok ? v : (half_t)0.f;
        }
        __syncthreads();

        // --- A fragment straight from global (L2-hot), ISA 16-bit A layout:
        //     lanes<16 K{0..7,16..23}, lanes>=16 K{8..15,24..31} ---
        Frag a;
        a.q[0] = *(const uint4*)(arow + kc * KCH_ + hi * 8);
        a.q[1] = *(const uint4*)(arow + kc * KCH_ + 16 + hi * 8);

        // --- B fragments (lanes<16: K{0..7,16..23}, lanes>=16: K{8..15,24..31}) + WMMA ---
        Frag bf;
        const uint4* bp;
#define DO_SUB(ACC, SUB)                                                      \
        bp = (const uint4*)(&Bs[((SUB) * 16 + lan) * BSTR_ + hi * 8]);        \
        bf.q[0] = bp[0]; bf.q[1] = bp[2];                                     \
        ACC = __builtin_amdgcn_wmma_f32_16x16x32_f16(false, a.v, false, bf.v, \
                                                     (short)0, ACC, false, false);
        DO_SUB(acc0, 0)
        DO_SUB(acc1, 1)
        DO_SUB(acc2, 2)
        DO_SUB(acc3, 3)
#undef DO_SUB
    }

    // --- store with channel shuffle: c_final = cout_head*HEADS + head ---
    // C/D layout: lanes 0-15 -> M=r, N=lane ; lanes 16-31 -> M=r+8, N=lane-16
    float* ob = out + (size_t)b * COUT_ * NTOT_;
    for (int r = 0; r < 8; ++r) {
        int M  = r + 8 * hi;
        int cf = (wave * 16 + M) * HEADS_ + head;
        float* orow = ob + (size_t)cf * NTOT_ + n0 + lan;
        orow[0]  = acc0[r];
        orow[16] = acc1[r];
        orow[32] = acc2[r];
        orow[48] = acc3[r];
    }
}

// ---------------------------------------------------------------------------
// Launch
// ---------------------------------------------------------------------------
extern "C" void kernel_launch(void* const* d_in, const int* in_sizes, int n_in,
                              void* d_out, int out_size, void* d_ws, size_t ws_size,
                              hipStream_t stream)
{
    const float* x     = (const float*)d_in[0];
    const float* gamma = (const float*)d_in[1];
    const float* beta  = (const float*)d_in[2];
    const float* mean  = (const float*)d_in[3];
    const float* var   = (const float*)d_in[4];
    const float* fc1   = (const float*)d_in[5];
    const float* fc2   = (const float*)d_in[6];
    const float* fc2b  = (const float*)d_in[7];
    const float* convw = (const float*)d_in[8];
    const int*   inact = (const int*)d_in[9];
    float* out = (float*)d_out;

    // Workspace layout (all offsets 16B-aligned)
    char* ws = (char*)d_ws;
    half_t* xr    = (half_t*)(ws);                      // 18,874,368 B
    half_t* Apack = (half_t*)(ws + 18874368);           //  4,128,768 B
    float*  xa    = (float*) (ws + 23003136);           //      1,536 B
    float*  maskw = (float*) (ws + 23004672);           //      6,144 B
    int*    actL  = (int*)   (ws + 23010816);           //      6,144 B
    int*    actC  = (int*)   (ws + 23016960);           //         64 B
    int2*   tabG  = (int2*)  (ws + 23017024);           //    344,064 B

    k_bn_relu_pool<<<dim3(B_ * C_), dim3(256), 0, stream>>>(
        x, gamma, beta, mean, var, xr, xa);

    k_gate<<<dim3(B_ * HEADS_), dim3(128), 0, stream>>>(
        xa, fc1, fc2, fc2b, inact, maskw, actL, actC);

    int nA = B_ * HEADS_ * MH_ * KTAB_;   // 2,064,384
    k_prepA<<<dim3((nA + 255) / 256), dim3(256), 0, stream>>>(
        convw, maskw, actL, actC, Apack);

    int nT = B_ * HEADS_ * KTAB_;         // 43,008
    k_prepT<<<dim3((nT + 255) / 256), dim3(256), 0, stream>>>(
        actL, actC, tabG);

    k_conv<<<dim3(NTOT_ / NTILE_, HEADS_, B_), dim3(96), 0, stream>>>(
        xr, Apack, tabG, actC, out);
}
